// VectorQuantizedVAE_71305047048237
// MI455X (gfx1250) — matrix operand order repaired and checked by reading, not compiled, not run on hardware
//
#include <hip/hip_runtime.h>
#include <hip/hip_bf16.h>

// ---------------------------------------------------------------------------
// VQ-VAE quantize for MI455X (gfx1250): bf16 WMMA scoring + fp32 gather.
// Round 3: fix async builtin pointer types (v4i as(1)/as(3) pointers).
// ---------------------------------------------------------------------------

typedef __attribute__((ext_vector_type(16))) __bf16 v16bf;
typedef __attribute__((ext_vector_type(8)))  float  v8f;

#define DDIM   256      // embedding dim
#define KCODES 512      // codebook size
#define MT     128      // rows (spatial positions) per workgroup
#define KC     128      // codebook chunk (codes resident in LDS at once)
#define PITCH  264      // ushort pitch per row (256 + 8 pad; 528B, 16B aligned)
#define HWT    1024     // H*W
#define NOUT   16777216 // 64*256*32*32

#if defined(__has_builtin)
#if __has_builtin(__builtin_amdgcn_global_load_async_to_lds_b128) && \
    __has_builtin(__builtin_amdgcn_s_wait_asynccnt)
#define USE_ASYNC_LDS 1
#endif
#endif

__device__ __forceinline__ unsigned short f2bf(float f) {
    unsigned u = __builtin_bit_cast(unsigned, f);
    u += 0x7FFFu + ((u >> 16) & 1u);           // round-to-nearest-even
    return (unsigned short)(u >> 16);
}

struct U4x2 { uint4 lo; uint4 hi; };
__device__ __forceinline__ v16bf make_frag(uint4 lo, uint4 hi) {
    U4x2 t; t.lo = lo; t.hi = hi;
    return __builtin_bit_cast(v16bf, t);
}

#ifdef USE_ASYNC_LDS
// Builtin signature (from hipcc diagnostic): param0 = int4-vector pointer in
// global AS, param1 = int4-vector pointer in LDS AS, then imm offset, imm cpol.
typedef int v4i __attribute__((vector_size(16)));
typedef __attribute__((address_space(1))) v4i* g_v4i_p;
typedef __attribute__((address_space(3))) v4i* l_v4i_p;

__device__ __forceinline__ void async_cp16(const unsigned short* gsrc,
                                           unsigned short* ldst) {
    // reinterpret in generic space first, then address-space-convert
    v4i* g_gen = (v4i*)(void*)const_cast<unsigned short*>(gsrc);
    v4i* l_gen = (v4i*)(void*)ldst;
    __builtin_amdgcn_global_load_async_to_lds_b128(
        (g_v4i_p)g_gen, (l_v4i_p)l_gen, 0, 0);
}
#endif

// ---------------------------------------------------------------------------
// Prep: emb (fp32) -> bf16 codebook in ws, plus halfnorm[k] = 0.5*||e_k||^2
// ---------------------------------------------------------------------------
__global__ __launch_bounds__(256) void vq_prep(const float* __restrict__ emb,
                                               unsigned short* __restrict__ cb,
                                               float* __restrict__ hnorm) {
    __shared__ float red[256];
    const int k = blockIdx.x;      // 512 blocks
    const int d = threadIdx.x;     // 256 threads
    float v = emb[k * DDIM + d];
    cb[k * DDIM + d] = f2bf(v);
    red[d] = v * v;
    __syncthreads();
    for (int s = 128; s > 0; s >>= 1) {
        if (d < s) red[d] += red[d + s];
        __syncthreads();
    }
    if (d == 0) hnorm[k] = 0.5f * red[0];
}

// ---------------------------------------------------------------------------
// Main: per-WG tile of 128 rows vs all 512 codes, bf16 WMMA, argmax, gather.
// ---------------------------------------------------------------------------
__global__ __launch_bounds__(256) void vq_main(const float* __restrict__ x,
                                               const float* __restrict__ emb,
                                               const unsigned short* __restrict__ cb,
                                               const float* __restrict__ hnorm,
                                               float* __restrict__ out0,
                                               float* __restrict__ out1,
                                               float* __restrict__ out2) {
    extern __shared__ unsigned short smem[];
    unsigned short* zl   = smem;                       // MT * PITCH ushorts
    unsigned short* el   = smem + MT * PITCH;          // KC * PITCH ushorts
    int*            idxl = (int*)(smem + (MT + KC) * PITCH);

    const int tid  = threadIdx.x;
    const int wave = tid >> 5;
    const int lane = tid & 31;

    const int    n0    = blockIdx.x * MT;
    const int    b     = n0 >> 10;
    const int    hw0   = n0 & (HWT - 1);
    const size_t xbase = (size_t)b * DDIM * HWT + hw0;

    // --- prefetch codebook toward cache while staging z ---
    for (int e = tid; e < (KCODES * DDIM * 2) / 128; e += 256)
        __builtin_prefetch((const char*)cb + e * 128, 0, 1);

    // --- stage z tile: fp32 global (d-major, hw contiguous) -> bf16 LDS row-major ---
    for (int e = tid; e < MT * DDIM / 4; e += 256) {
        int d  = e >> 5;            // 0..255
        int i4 = e & 31;            // 0..31 -> rows i4*4..i4*4+3
        float4 v = *(const float4*)(x + xbase + (size_t)d * HWT + i4 * 4);
        int i = i4 * 4;
        zl[(i + 0) * PITCH + d] = f2bf(v.x);
        zl[(i + 1) * PITCH + d] = f2bf(v.y);
        zl[(i + 2) * PITCH + d] = f2bf(v.z);
        zl[(i + 3) * PITCH + d] = f2bf(v.w);
    }
    __syncthreads();

    // --- load this wave's A fragments (16 rows x 256 K) into registers ---
    // ISA layout 16-bit A 16x32: lanes 0-15 hold K[0..7],[16..23]; lanes 16-31 K[8..15],[24..31]
    v16bf afrag[8];
    {
        const int row  = wave * 16 + (lane & 15);
        const int dsel = (lane < 16) ? 0 : 8;
        const unsigned short* zp = zl + row * PITCH + dsel;
        #pragma unroll
        for (int ks = 0; ks < 8; ++ks) {
            uint4 lo = *(const uint4*)(zp + ks * 32);
            uint4 hi = *(const uint4*)(zp + ks * 32 + 16);
            afrag[ks] = make_frag(lo, hi);
        }
    }

    float bval[8];
    int   bidx[8];
    #pragma unroll
    for (int j = 0; j < 8; ++j) { bval[j] = -3.4e38f; bidx[j] = 0; }

    // --- sweep codebook in LDS-resident chunks ---
    for (int chunk = 0; chunk < KCODES / KC; ++chunk) {
        __syncthreads();  // previous chunk's B reads done before overwrite
#ifdef USE_ASYNC_LDS
        // ASYNCcnt-tracked direct global->LDS copy (no VGPR round trip)
        for (int e = tid; e < KC * (DDIM / 8); e += 256) {
            int k = e >> 5;         // code within chunk
            int j = e & 31;         // 16B segment within row
            async_cp16(cb + (chunk * KC + k) * DDIM + j * 8,
                       el + k * PITCH + j * 8);
        }
        __builtin_amdgcn_s_wait_asynccnt(0);
#else
        for (int e = tid; e < KC * (DDIM / 8); e += 256) {
            int k = e >> 5;
            int j = e & 31;
            *(uint4*)(el + k * PITCH + j * 8) =
                *(const uint4*)(cb + (chunk * KC + k) * DDIM + j * 8);
        }
#endif
        __syncthreads();

        // two n-tiles in flight: independent accumulators give the scheduler
        // 4 ds_load_b128 + 2 WMMAs per k-step to overlap
        for (int nt = 0; nt < KC / 16; nt += 2) {
            v8f acc0 = {};
            v8f acc1 = {};
            // ISA layout 16-bit B 32x16: lane = column(code), lanes 0-15 K[0..7],[16..23]; lanes 16-31 K[8..15],[24..31]
            const unsigned short* ep0 =
                el + (nt * 16 + (lane & 15)) * PITCH + (lane >> 4) * 8;
            const unsigned short* ep1 = ep0 + 16 * PITCH;
            #pragma unroll
            for (int ks = 0; ks < 8; ++ks) {
                uint4 l0 = *(const uint4*)(ep0 + ks * 32);
                uint4 h0 = *(const uint4*)(ep0 + ks * 32 + 16);
                uint4 l1 = *(const uint4*)(ep1 + ks * 32);
                uint4 h1 = *(const uint4*)(ep1 + ks * 32 + 16);
                acc0 = __builtin_amdgcn_wmma_f32_16x16x32_bf16(
                    false, afrag[ks], false, make_frag(l0, h0), (short)0, acc0, false, false);
                acc1 = __builtin_amdgcn_wmma_f32_16x16x32_bf16(
                    false, afrag[ks], false, make_frag(l1, h1), (short)0, acc1, false, false);
            }
            const int   code0 = chunk * KC + nt * 16 + (lane & 15);
            const int   code1 = code0 + 16;
            const float hn0   = hnorm[code0];
            const float hn1   = hnorm[code1];
            #pragma unroll
            for (int j = 0; j < 8; ++j) {
                float s0 = acc0[j] - hn0;  // score = z.e - 0.5||e||^2 (argmax == argmin dist)
                bool b0 = (s0 > bval[j]) || (s0 == bval[j] && code0 < bidx[j]);
                if (b0) { bval[j] = s0; bidx[j] = code0; }
                float s1 = acc1[j] - hn1;
                bool b1 = (s1 > bval[j]) || (s1 == bval[j] && code1 < bidx[j]);
                if (b1) { bval[j] = s1; bidx[j] = code1; }
            }
        }
    }

    // --- argmax reduce across the 16 lanes of each column group ---
    #pragma unroll
    for (int j = 0; j < 8; ++j) {
        float v  = bval[j];
        int   ii = bidx[j];
        #pragma unroll
        for (int off = 8; off >= 1; off >>= 1) {
            float ov = __shfl_xor(v, off, 32);
            int   oi = __shfl_xor(ii, off, 32);
            if (ov > v || (ov == v && oi < ii)) { v = ov; ii = oi; }
        }
        const int m = j + ((lane >= 16) ? 8 : 0);   // C layout: vgpr j -> row j / j+8
        if ((lane & 15) == 0) idxl[wave * 16 + m] = ii;
    }
    __syncthreads();

    // --- gather fp32 codes + write (codes, x, codes), coalesced float4 stores ---
    for (int e = tid; e < DDIM * MT / 4; e += 256) {
        int d  = e >> 5;
        int i4 = e & 31;
        int i  = i4 * 4;
        size_t off = xbase + (size_t)d * HWT + i;
        float4 xv = *(const float4*)(x + off);
        float4 c;
        c.x = emb[idxl[i + 0] * DDIM + d];
        c.y = emb[idxl[i + 1] * DDIM + d];
        c.z = emb[idxl[i + 2] * DDIM + d];
        c.w = emb[idxl[i + 3] * DDIM + d];
        *(float4*)(out0 + off) = c;
        *(float4*)(out1 + off) = xv;
        *(float4*)(out2 + off) = c;
    }
}

// ---------------------------------------------------------------------------
extern "C" void kernel_launch(void* const* d_in, const int* in_sizes, int n_in,
                              void* d_out, int out_size, void* d_ws, size_t ws_size,
                              hipStream_t stream) {
    const float* x   = (const float*)d_in[0];   // [64,256,32,32]
    const float* emb = (const float*)d_in[1];   // [512,256]

    unsigned short* cb    = (unsigned short*)d_ws;                 // 512*256 bf16
    float*          hnorm = (float*)((char*)d_ws + KCODES * DDIM * 2);

    float* out0 = (float*)d_out;
    float* out1 = out0 + NOUT;
    float* out2 = out1 + NOUT;

    vq_prep<<<KCODES, 256, 0, stream>>>(emb, cb, hnorm);

    const size_t smem_bytes = (size_t)(MT + KC) * PITCH * 2 + MT * sizeof(int);
    vq_main<<<(64 * HWT) / MT, 256, smem_bytes, stream>>>(
        x, emb, cb, hnorm, out0, out1, out2);
}
